// MoELayer_2276332667120
// MI455X (gfx1250) — matrix operand order repaired and checked by reading, not compiled, not run on hardware
//
#include <hip/hip_runtime.h>
#include <hip/hip_bf16.h>

#define T_TOK  4096
#define DDIM   1024
#define HDIM   4096
#define EDIM   1024
#define NEXP   8
#define M_TILE 32
#define HCHUNK 256

typedef __attribute__((ext_vector_type(16))) __bf16          v16bf;
typedef __attribute__((ext_vector_type(16))) unsigned short  v16us;
typedef __attribute__((ext_vector_type(8)))  float           v8f;

union FragU { uint4 q[2]; v16us s; };

__device__ __forceinline__ unsigned short f2bf(float f) {
  unsigned u = __builtin_bit_cast(unsigned, f);
  unsigned r = u + 0x7FFFu + ((u >> 16) & 1u);   // round-to-nearest-even
  return (unsigned short)(r >> 16);
}
__device__ __forceinline__ unsigned pack2(float a, float b) {
  return (unsigned)f2bf(a) | ((unsigned)f2bf(b) << 16);
}
__device__ __forceinline__ v8f vzero8() {
  v8f z = {0.f, 0.f, 0.f, 0.f, 0.f, 0.f, 0.f, 0.f};
  return z;
}

// ---------------------------------------------------------------------------
// Kernel 0: zero output accumulator + routing counters
// ---------------------------------------------------------------------------
__global__ void zero_kernel(float* __restrict__ out, int n,
                            int* __restrict__ cnt, float* __restrict__ imp,
                            float* __restrict__ ldacc) {
  int i = blockIdx.x * blockDim.x + threadIdx.x;
  for (; i < n; i += gridDim.x * blockDim.x) out[i] = 0.f;
  if (blockIdx.x == 0 && threadIdx.x < NEXP) {
    cnt[threadIdx.x]   = 0;
    imp[threadIdx.x]   = 0.f;
    ldacc[threadIdx.x] = 0.f;
  }
}

// ---------------------------------------------------------------------------
// Kernel 1: convert fp32 weights -> bf16, pre-swizzled into this code's contiguous-k WMMA
// B-fragment layout: dst[((e*(K/32) + k/32)*NC + n)*32 + (k%32)]  (ushort)
// A lane's 16 B-frag elements (k = kb + half*16 .. +16, fixed n) become one
// contiguous 32-byte run -> two global_load_b128 in the GEMM.
// ---------------------------------------------------------------------------
#define CONV_COLS 256
#define LSPAD 48   // row pitch (ushorts): 96B, multiple of 16B for b128 access

__global__ void convert_swizzle_kernel(const float* __restrict__ src,
                                       unsigned short* __restrict__ dst,
                                       int K, int NC) {
  __shared__ __align__(16) unsigned short Ls[CONV_COLS * LSPAD];  // 24 KB

  const int blocksPerE = (K >> 5) * (NC / CONV_COLS);
  const int e    = blockIdx.x / blocksPerE;
  const int rem  = blockIdx.x % blocksPerE;
  const int kb32 = rem / (NC / CONV_COLS);
  const int cc   = rem % (NC / CONV_COLS);
  const int kbase   = kb32 << 5;
  const int colbase = cc * CONV_COLS;
  const int tid = threadIdx.x;

  // phase 1: coalesced fp32 read (over columns), bf16 convert, transpose to LDS
  for (int i = tid; i < 32 * CONV_COLS; i += 256) {
    int r = i >> 8;          // k within block
    int c = i & (CONV_COLS - 1);
    float f = src[((size_t)e * K + kbase + r) * NC + colbase + c];
    Ls[c * LSPAD + r] = f2bf(f);
  }
  __syncthreads();

  // phase 2: each thread emits one 32-byte swizzled chunk (col, half)
  for (int it = 0; it < 2; it++) {
    int idx  = tid + it * 256;          // 512 chunks = 256 cols x 2 halves
    int col  = idx >> 1;
    int half = idx & 1;
    const uint4* s = (const uint4*)&Ls[col * LSPAD + half * 16];
    size_t dbase = (((size_t)e * (K >> 5) + kb32) * NC + colbase + col) * 32
                   + half * 16;
    uint4* d = (uint4*)&dst[dbase];
    d[0] = s[0];
    d[1] = s[1];
  }
}

// ---------------------------------------------------------------------------
// Kernel 2: gating. One wave32 per token: logits, softmax, top-2 routing.
// ---------------------------------------------------------------------------
__global__ void gate_kernel(const float* __restrict__ x,
                            const float* __restrict__ Wg,
                            const float* __restrict__ bg,
                            int* __restrict__ cnt, float* __restrict__ imp,
                            float* __restrict__ ldacc,
                            int* __restrict__ list, float* __restrict__ rgate) {
  const int wave = threadIdx.x >> 5;
  const int lane = threadIdx.x & 31;
  const int t = blockIdx.x * 4 + wave;

  float acc[NEXP];
#pragma unroll
  for (int n = 0; n < NEXP; n++) acc[n] = 0.f;

  const float* xr = x + (size_t)t * DDIM;
  for (int d = lane; d < DDIM; d += 32) {
    float xv = xr[d];
    const float* wr = Wg + (size_t)d * NEXP;
#pragma unroll
    for (int n = 0; n < NEXP; n++) acc[n] = fmaf(xv, wr[n], acc[n]);
  }
#pragma unroll
  for (int n = 0; n < NEXP; n++) {
#pragma unroll
    for (int off = 16; off >= 1; off >>= 1)
      acc[n] += __shfl_xor(acc[n], off, 32);
  }

  if (lane == 0) {
    float p[NEXP];
    float mx = -1e30f;
#pragma unroll
    for (int n = 0; n < NEXP; n++) {
      p[n] = acc[n] + bg[n];
      mx = fmaxf(mx, p[n]);
    }
    float s = 0.f;
#pragma unroll
    for (int n = 0; n < NEXP; n++) { p[n] = __expf(p[n] - mx); s += p[n]; }
    float inv = 1.f / s;
#pragma unroll
    for (int n = 0; n < NEXP; n++) p[n] *= inv;

    int i1 = 0;
#pragma unroll
    for (int n = 1; n < NEXP; n++) if (p[n] > p[i1]) i1 = n;
    int i2 = (i1 == 0) ? 1 : 0;
#pragma unroll
    for (int n = 0; n < NEXP; n++) if (n != i1 && p[n] > p[i2]) i2 = n;

    float denom = fmaxf(p[i1] + p[i2], 1e-12f);
    float g1 = p[i1] / denom, g2 = p[i2] / denom;

#pragma unroll
    for (int n = 0; n < NEXP; n++) atomicAdd(&imp[n], p[n]);
    atomicAdd(&ldacc[i1], 1.f);
    atomicAdd(&ldacc[i2], 1.f);

    int s1 = atomicAdd(&cnt[i1], 1);
    list[i1 * T_TOK + s1] = t;  rgate[i1 * T_TOK + s1] = g1;
    int s2 = atomicAdd(&cnt[i2], 1);
    list[i2 * T_TOK + s2] = t;  rgate[i2 * T_TOK + s2] = g2;
  }
}

// ---------------------------------------------------------------------------
// Kernel 3: routed expert FFN, bf16 WMMA, swizzled bf16 weights.
// Block = (expert, 32-token tile), 8 waves; wave owns 128 output columns.
// ---------------------------------------------------------------------------
__launch_bounds__(256, 1)
__global__ void moe_ffn_kernel(const float* __restrict__ x,
                               const unsigned short* __restrict__ W1s,
                               const float* __restrict__ b1,
                               const unsigned short* __restrict__ W2s,
                               const float* __restrict__ b2,
                               const int* __restrict__ cnt,
                               const int* __restrict__ list,
                               const float* __restrict__ rgate,
                               float* __restrict__ out) {
  __shared__ __align__(16) unsigned short Xs[M_TILE * DDIM];    // 64 KB bf16 X
  __shared__ __align__(16) unsigned short Hs[M_TILE * HCHUNK];  // 16 KB bf16 H
  __shared__ int   tokS[M_TILE];
  __shared__ float gS[M_TILE];

  const int e       = blockIdx.y;
  const int count   = cnt[e];
  const int rowbase = blockIdx.x * M_TILE;
  if (rowbase >= count) return;   // uniform per block

  const int tid  = threadIdx.x;
  const int wave = tid >> 5;
  const int lane = tid & 31;
  const int half = lane >> 4;
  const int mn   = lane & 15;

  if (tid < M_TILE) {
    int idx = rowbase + tid;
    int tok = 0; float g = 0.f;
    if (idx < count) { tok = list[e * T_TOK + idx]; g = rgate[e * T_TOK + idx]; }
    tokS[tid] = tok;  gS[tid] = g;
  }
  __syncthreads();

  // gather X tile: float4 loads, bf16 convert, row-major LDS
  for (int i4 = tid; i4 < M_TILE * (DDIM / 4); i4 += 256) {
    int r  = i4 >> 8;              // / (DDIM/4)
    int c4 = i4 & (DDIM / 4 - 1);
    float4 v = ((const float4*)(x + (size_t)tokS[r] * DDIM))[c4];
    unsigned lo = pack2(v.x, v.y);
    unsigned hi = pack2(v.z, v.w);
    *(uint2*)&Xs[r * DDIM + c4 * 4] = make_uint2(lo, hi);
  }
  __syncthreads();

  // swizzled bf16 weight bases for this expert
  const unsigned short* W1e = W1s + (size_t)e * (DDIM >> 5) * HDIM * 32;
  const unsigned short* W2e = W2s + (size_t)e * (HDIM >> 5) * EDIM * 32;
  const float* b1e = b1 + (size_t)e * HDIM;
  const float* b2e = b2 + (size_t)e * EDIM;

  // persistent f32 accumulators: 32 rows x 128 cols per wave
  v8f occ[2][8];
#pragma unroll
  for (int rt = 0; rt < 2; rt++)
#pragma unroll
    for (int ct = 0; ct < 8; ct++) occ[rt][ct] = vzero8();

#pragma unroll 1
  for (int hit = 0; hit < HDIM / HCHUNK; hit++) {
    const int hbase = hit * HCHUNK;
    const int hw    = hbase + wave * 32;   // this wave's 32 hidden columns

    // ---- GEMM1: Hwave[32x32] = relu(X[32x1024] @ W1[:, hw..hw+32) + b1)
    v8f acc1[2][2];
#pragma unroll
    for (int rt = 0; rt < 2; rt++)
#pragma unroll
      for (int ct = 0; ct < 2; ct++) acc1[rt][ct] = vzero8();

#pragma unroll 1
    for (int kb = 0; kb < DDIM; kb += 32) {
      __builtin_prefetch(W1e + (((size_t)((kb + 32) >> 5)) * HDIM + hw) * 32, 0, 1);
      // A frags: two ds_load_b128 each (k = kb + half*16 .. +16, matches B swizzle)
      FragU af[2];
#pragma unroll
      for (int rt = 0; rt < 2; rt++) {
        int r = rt * 16 + mn;
        af[rt].q[0] = *(const uint4*)&Xs[r * DDIM + kb + half * 16];
        af[rt].q[1] = *(const uint4*)&Xs[r * DDIM + kb + half * 16 + 8];
      }
#pragma unroll
      for (int ct = 0; ct < 2; ct++) {
        int col = hw + ct * 16 + mn;
        // B frag: one contiguous 32B run in swizzled storage -> 2x b128
        const uint4* bp = (const uint4*)
            &W1e[(((size_t)(kb >> 5)) * HDIM + col) * 32 + half * 16];
        FragU bfr;
        bfr.q[0] = bp[0];
        bfr.q[1] = bp[1];
        v16bf bv = __builtin_bit_cast(v16bf, bfr.s);
#pragma unroll
        for (int rt = 0; rt < 2; rt++) {
          v16bf av = __builtin_bit_cast(v16bf, af[rt].s);
          acc1[rt][ct] = __builtin_amdgcn_wmma_f32_16x16x32_bf16(
              false, av, false, bv, (short)0, acc1[rt][ct], false, false);
        }
      }
    }

    // bias + relu, stage bf16 hidden chunk to LDS
#pragma unroll
    for (int ct = 0; ct < 2; ct++) {
      int col = hw + ct * 16 + mn;
      float bb = b1e[col];
#pragma unroll
      for (int rt = 0; rt < 2; rt++) {
#pragma unroll
        for (int v = 0; v < 8; v++) {
          int m = rt * 16 + half * 8 + v;   // C-tile row per ISA layout
          float hv = fmaxf(acc1[rt][ct][v] + bb, 0.f);
          Hs[m * HCHUNK + (wave * 32 + ct * 16 + mn)] = f2bf(hv);
        }
      }
    }
    __syncthreads();

    // ---- GEMM2: occ[32x128 per wave] += Hs[32x256] @ W2[hbase.., wave*128..)
#pragma unroll 1
    for (int kb = 0; kb < HCHUNK; kb += 32) {
      __builtin_prefetch(
          W2e + (((size_t)((hbase + kb + 32) >> 5)) * EDIM + wave * 128) * 32, 0, 1);
      FragU af[2];
#pragma unroll
      for (int rt = 0; rt < 2; rt++) {
        int r = rt * 16 + mn;
        af[rt].q[0] = *(const uint4*)&Hs[r * HCHUNK + kb + half * 16];
        af[rt].q[1] = *(const uint4*)&Hs[r * HCHUNK + kb + half * 16 + 8];
      }
      v16bf av0 = __builtin_bit_cast(v16bf, af[0].s);
      v16bf av1 = __builtin_bit_cast(v16bf, af[1].s);
#pragma unroll
      for (int ct = 0; ct < 8; ct++) {
        int col = wave * 128 + ct * 16 + mn;
        const uint4* bp = (const uint4*)
            &W2e[(((size_t)((hbase + kb) >> 5)) * EDIM + col) * 32 + half * 16];
        FragU bfr;
        bfr.q[0] = bp[0];
        bfr.q[1] = bp[1];
        v16bf bv = __builtin_bit_cast(v16bf, bfr.s);
        occ[0][ct] = __builtin_amdgcn_wmma_f32_16x16x32_bf16(
            false, av0, false, bv, (short)0, occ[0][ct], false, false);
        occ[1][ct] = __builtin_amdgcn_wmma_f32_16x16x32_bf16(
            false, av1, false, bv, (short)0, occ[1][ct], false, false);
      }
    }
    __syncthreads();
  }

  // ---- scatter: out[token] += g * (occ + b2)
#pragma unroll
  for (int ct = 0; ct < 8; ct++) {
    int col = wave * 128 + ct * 16 + mn;
    float bb = b2e[col];
#pragma unroll
    for (int rt = 0; rt < 2; rt++) {
#pragma unroll
      for (int v = 0; v < 8; v++) {
        int m = rt * 16 + half * 8 + v;
        float g = gS[m];
        if (g != 0.f)
          atomicAdd(&out[(size_t)tokS[m] * EDIM + col], g * (occ[rt][ct][v] + bb));
      }
    }
  }
}

// ---------------------------------------------------------------------------
// Kernel 4: lb_loss + normalized importance/load into the output tail
// ---------------------------------------------------------------------------
__global__ void finalize_kernel(const float* __restrict__ imp,
                                const float* __restrict__ ldacc,
                                float* __restrict__ out_tail) {
  if (threadIdx.x == 0) {
    const float invT = 1.0f / (float)T_TOK;
    float loss = 0.f;
    float im[NEXP], ld[NEXP];
#pragma unroll
    for (int n = 0; n < NEXP; n++) {
      im[n] = imp[n] * invT;
      ld[n] = ldacc[n] * invT;
      loss += im[n] * ld[n];
    }
    out_tail[0] = loss * (float)NEXP;
#pragma unroll
    for (int n = 0; n < NEXP; n++) {
      out_tail[1 + n] = im[n];
      out_tail[9 + n] = ld[n];
    }
  }
}

// ---------------------------------------------------------------------------
extern "C" void kernel_launch(void* const* d_in, const int* in_sizes, int n_in,
                              void* d_out, int out_size, void* d_ws, size_t ws_size,
                              hipStream_t stream) {
  const float* x  = (const float*)d_in[0];
  const float* Wg = (const float*)d_in[1];
  const float* bg = (const float*)d_in[2];
  const float* W1 = (const float*)d_in[3];
  const float* b1 = (const float*)d_in[4];
  const float* W2 = (const float*)d_in[5];
  const float* b2 = (const float*)d_in[6];
  float* out = (float*)d_out;

  char*  ws    = (char*)d_ws;
  int*   cnt   = (int*)(ws + 0);
  float* imp   = (float*)(ws + 64);
  float* ldacc = (float*)(ws + 128);
  int*   list  = (int*)(ws + 256);
  float* rgate = (float*)(ws + 256 + (size_t)NEXP * T_TOK * sizeof(int));
  // swizzled bf16 weights (64 MiB each), 1 MiB-aligned offsets
  unsigned short* W1s = (unsigned short*)(ws + (1u << 20));
  unsigned short* W2s = (unsigned short*)(ws + (1u << 20) + ((size_t)1 << 26));

  const int OUT_MAIN = T_TOK * EDIM;   // 4096 x 1024

  zero_kernel<<<1024, 256, 0, stream>>>(out, OUT_MAIN, cnt, imp, ldacc);

  // one-time per launch: fp32 -> bf16 swizzled weights (L2-resident afterwards)
  convert_swizzle_kernel<<<NEXP * (DDIM >> 5) * (HDIM / CONV_COLS), 256, 0, stream>>>(
      W1, W1s, DDIM, HDIM);
  convert_swizzle_kernel<<<NEXP * (HDIM >> 5) * (EDIM / CONV_COLS), 256, 0, stream>>>(
      W2, W2s, HDIM, EDIM);

  gate_kernel<<<T_TOK / 4, 128, 0, stream>>>(x, Wg, bg, cnt, imp, ldacc, list, rgate);

  dim3 grid(T_TOK / M_TILE, NEXP);     // worst case: all tokens on one expert
  moe_ffn_kernel<<<grid, 256, 0, stream>>>(x, W1s, b1, W2s, b2, cnt, list, rgate, out);

  finalize_kernel<<<1, 32, 0, stream>>>(imp, ldacc, out + OUT_MAIN);
}
